// DialogueGAT_17059610100341
// MI455X (gfx1250) — matrix operand order repaired and checked by reading, not compiled, not run on hardware
//
#include <hip/hip_runtime.h>
#include <hip/hip_bf16.h>

typedef __attribute__((ext_vector_type(16))) _Float16 v16h;
typedef __attribute__((ext_vector_type(8)))  _Float16 v8h;
typedef __attribute__((ext_vector_type(8)))  float    v8f;

#define D_   300   // feature dim
#define DP   320   // K-padded feature dim (multiple of 32)
#define NT_  2304  // total graph nodes  (32 * 72)
#define NF   1504  // N-padded H*D (1500 -> 1504)
#define NU   2048  // utterances (32*64)
#define LSEQ 256   // sentence length
#define LR   260   // LDS rows (L + max_w - 1, padded with zeros)
#define HH   5     // heads
#define BB   32    // dialogues
#define UU   64    // utterances per dialogue
#define PP   8     // parties per dialogue
#define NODE_STRIDE 72
#define KSTEPS (DP / 32)       // 10
#define FRAG   512             // halves per packed 32x16 B tile (32 lanes * 16)
#define CNT    7               // conv filter N-tiles (112/16)
#define GNT    (NF / 16)       // 94 GAT N-tiles

#define CDIV(a,b) (((a)+(b)-1)/(b))

// ---------------- WMMA fragment helpers (ISA 7.12.2 layouts) ----------------

// 16-bit A matrix 16x32: lanes 0-15 hold K=0..7,16..23; lanes 16-31 hold K=8..15,24..31
// two aligned 16B loads per lane (ds_load_b128 / global_load_b128)
__device__ inline v16h load_a_frag(const _Float16* __restrict__ base, int stride, int lane) {
  int m  = lane & 15;
  int kb = (lane < 16) ? 0 : 8;
  const _Float16* p = base + m * stride + kb;
  v8h lo = *(const v8h*)(p);
  v8h hi = *(const v8h*)(p + 16);
  return __builtin_shufflevector(lo, hi, 0, 1, 2, 3, 4, 5, 6, 7,
                                 8, 9, 10, 11, 12, 13, 14, 15);
}

// B tile pre-packed in fragment order: lane l's 16 halves contiguous -> one v16h load
__device__ inline v16h load_b_frag_packed(const _Float16* __restrict__ tile, int lane) {
  return *(const v16h*)(tile + lane * 16);
}

// 32-bit C/D 16x16: VGPR j holds M=j (lanes 0-15) / M=8+j (lanes 16-31), N = lane&15
__device__ inline void store_c_frag(float* __restrict__ base, int stride, int lane, v8f c) {
  int n  = lane & 15;
  int mb = (lane < 16) ? 0 : 8;
#pragma unroll
  for (int j = 0; j < 8; ++j) base[(size_t)(mb + j) * stride + n] = c[j];
}

// float atomic-max via CAS (works for LDS and global pointers)
__device__ inline void atomic_fmax(float* addr, float v) {
  unsigned int* ia = reinterpret_cast<unsigned int*>(addr);
  unsigned int old = *ia;
  while (v > __uint_as_float(old)) {
    unsigned int assumed = old;
    old = atomicCAS(ia, assumed, __float_as_uint(v));
    if (old == assumed) break;
  }
}

__device__ inline float block_reduce_sum(float v, float* red) {
  int tid = threadIdx.x;
  red[tid] = v;
  __syncthreads();
  for (int s = 128; s > 0; s >>= 1) {
    if (tid < s) red[tid] += red[tid + s];
    __syncthreads();
  }
  float r = red[0];
  __syncthreads();
  return r;
}

// ---------------- weight packing (fragment-order B tiles) ----------------
// decode a packed element index within one [ntiles][KSTEPS][FRAG] slice
__device__ inline void decode_frag_idx(int rem, int* kOut, int* nOut, int* ntOut, int* kkOut) {
  int nt  = rem / (KSTEPS * FRAG);
  int r3  = rem - nt * (KSTEPS * FRAG);
  int kk  = r3 / FRAG;
  int li  = r3 - kk * FRAG;
  int lane = li >> 4;
  int j    = li & 15;
  *kOut  = kk * 32 + ((lane < 16) ? 0 : 8) + ((j < 8) ? j : 8 + j);
  *nOut  = nt * 16 + (lane & 15);
  *ntOut = nt;
  *kkOut = kk;
}

// conv filters -> packed f16 B tiles, layout [12 r-slices][CNT][KSTEPS][FRAG]
__global__ void pack_filters_kernel(const float* __restrict__ w3, const float* __restrict__ w4,
                                    const float* __restrict__ w5, _Float16* __restrict__ wfilt) {
  int idx = blockIdx.x * blockDim.x + threadIdx.x;
  const int slice = CNT * KSTEPS * FRAG;
  const int total = 12 * slice;
  if (idx >= total) return;
  int s   = idx / slice;
  int rem = idx - s * slice;
  int k, n, nt, kk;
  decode_frag_idx(rem, &k, &n, &nt, &kk);
  float v = 0.f;
  if (k < D_ && n < 100) {
    if (s < 3)      v = w3[n * (3 * D_) + s       * D_ + k];
    else if (s < 7) v = w4[n * (4 * D_) + (s - 3) * D_ + k];
    else            v = w5[n * (5 * D_) + (s - 7) * D_ + k];
  }
  wfilt[idx] = (_Float16)v;
}

// GAT fc/res weights -> packed f16 B tiles, layout [5][2][GNT][KSTEPS][FRAG]
__global__ void pack_gatw_kernel(const float* __restrict__ fc, const float* __restrict__ res,
                                 _Float16* __restrict__ gw) {
  int idx = blockIdx.x * blockDim.x + threadIdx.x;
  const int slice = GNT * KSTEPS * FRAG;
  const int total = 5 * 2 * slice;
  if (idx >= total) return;
  int s     = idx / (2 * slice);
  int rem   = idx - s * (2 * slice);
  int which = rem / slice;
  int r2    = rem - which * slice;
  int k, n, nt, kk;
  decode_frag_idx(r2, &k, &n, &nt, &kk);
  float v = 0.f;
  if (k < D_ && n < 1500) {
    const float* srcw = (which == 0) ? fc : res;
    v = srcw[s * D_ * 1500 + k * 1500 + n];
  }
  gw[idx] = (_Float16)v;
}

__global__ void zero_feat_kernel(float* __restrict__ hf32, _Float16* __restrict__ hf16) {
  int idx = blockIdx.x * blockDim.x + threadIdx.x;
  if (idx < NT_ * DP) hf16[idx] = (_Float16)0.f;
  if (idx < NT_ * D_) hf32[idx] = 0.f;
}

__global__ void party_fill_kernel(const int* __restrict__ q_idx, const int* __restrict__ pids,
                                  const float* __restrict__ party_W,
                                  float* __restrict__ hf32, _Float16* __restrict__ hf16) {
  int idx = blockIdx.x * blockDim.x + threadIdx.x;
  if (idx >= BB * PP * D_) return;
  int j = idx / D_;
  int d = idx - j * D_;
  int node = q_idx[j];
  float v = party_W[pids[j] * D_ + d];
  hf32[(size_t)node * D_ + d] = v;
  hf16[(size_t)node * DP + d] = (_Float16)v;
}

// ---------------- TextCNN: LDS-staged implicit GEMM via WMMA ----------------
// one workgroup (8 wave32) per utterance; LDS holds the f16-embedded sentence
__global__ void __launch_bounds__(256)
conv_embed_kernel(const int* __restrict__ x, const float* __restrict__ word_W,
                  const _Float16* __restrict__ wfilt,
                  const float* __restrict__ b3, const float* __restrict__ b4,
                  const float* __restrict__ b5, const float* __restrict__ pos_W,
                  float* __restrict__ hf32, _Float16* __restrict__ hf16) {
  extern __shared__ char smem[];
  _Float16* emb    = (_Float16*)smem;                                       // [LR][DP]
  float*    maxbuf = (float*)(smem + (size_t)LR * DP * sizeof(_Float16));   // [3*112]
  const int n   = blockIdx.x;
  const int tid = threadIdx.x;

  if (tid < 3 * 112) maxbuf[tid] = -1e30f;

  // gather + f32->f16 convert into LDS, zero padding (cols >= 300, rows >= 256)
  for (int idx = tid; idx < LR * DP; idx += 256) {
    int row = idx / DP;
    int col = idx - row * DP;
    float v = 0.f;
    if (row < LSEQ && col < D_) {
      int tok = x[n * LSEQ + row];
      v = word_W[(size_t)tok * D_ + col];
    }
    emb[idx] = (_Float16)v;
  }
  __syncthreads();

  const int wave = tid >> 5, lane = tid & 31;
  // combos: 3 widths x 16 position-tiles x 7 filter-tiles
  for (int combo = wave; combo < 3 * 16 * CNT; combo += 8) {
    int wi  = combo / (16 * CNT);
    int rem = combo - wi * (16 * CNT);
    int mt  = rem / CNT;
    int nt  = rem - mt * CNT;
    int w   = 3 + wi;
    int s0  = (wi == 0) ? 0 : (wi == 1 ? 3 : 7);
    v8f acc = {};
    for (int r = 0; r < w; ++r) {
      const _Float16* Btiles = wfilt + (size_t)((s0 + r) * CNT + nt) * KSTEPS * FRAG;
      const _Float16* Ab     = emb + (size_t)(mt * 16 + r) * DP;
      for (int kk = 0; kk < KSTEPS; ++kk) {
        v16h a = load_a_frag(Ab + kk * 32, DP, lane);
        v16h b = load_b_frag_packed(Btiles + kk * FRAG, lane);
        acc = __builtin_amdgcn_wmma_f32_16x16x32_f16(false, a, false, b,
                                                     (short)0, acc, false, false);
      }
    }
    // per-filter max over the tile's 16 positions (mask invalid tail positions)
    int nn = lane & 15;
    int mb = (lane < 16) ? 0 : 8;
    int plimit = LSEQ - w + 1;
    float mymax = -1e30f;
#pragma unroll
    for (int j = 0; j < 8; ++j) {
      int p = mt * 16 + mb + j;
      float vv = acc[j];
      if (p < plimit && vv > mymax) mymax = vv;
    }
    float other = __shfl_xor(mymax, 16);
    if (other > mymax) mymax = other;
    if (lane < 16 && mymax > -1e29f)
      atomic_fmax(&maxbuf[wi * 112 + nt * 16 + nn], mymax);
  }
  __syncthreads();

  // write wx + bias + positional embedding into node features
  int g = n >> 6, iu = n & 63;
  int node = g * NODE_STRIDE + iu;
  for (int col = tid; col < D_; col += 256) {
    int wi = col / 100;
    int f  = col - wi * 100;
    const float* bp = (wi == 0) ? b3 : (wi == 1 ? b4 : b5);
    float val = maxbuf[wi * 112 + f] + bp[f] + pos_W[iu * D_ + col];
    hf32[(size_t)node * D_ + col] = val;
    hf16[(size_t)node * DP + col] = (_Float16)val;
  }
}

// -------- generic WMMA GEMM: C[M x N] = A[M x K] * Bpacked[K x N] --------
__global__ void __launch_bounds__(128)
wmma_gemm_kernel(const _Float16* __restrict__ A, const _Float16* __restrict__ Bm,
                 float* __restrict__ C, int mtiles, int ntiles, int ksteps,
                 int lda, int ldc) {
  int wave = threadIdx.x >> 5, lane = threadIdx.x & 31;
  int tile = blockIdx.x * 4 + wave;
  if (tile >= mtiles * ntiles) return;
  int mt = tile / ntiles, nt = tile - mt * ntiles;
  const _Float16* Abase  = A + (size_t)mt * 16 * lda;
  const _Float16* Btiles = Bm + (size_t)nt * ksteps * FRAG;
  v8f acc = {};
  for (int kk = 0; kk < ksteps; ++kk) {
    v16h a = load_a_frag(Abase + kk * 32, lda, lane);
    v16h b = load_b_frag_packed(Btiles + kk * FRAG, lane);
    acc = __builtin_amdgcn_wmma_f32_16x16x32_f16(false, a, false, b,
                                                 (short)0, acc, false, false);
  }
  store_c_frag(C + (size_t)mt * 16 * ldc + nt * 16, ldc, lane, acc);
}

// ---------------- GAT graph ops ----------------

__global__ void el_er_kernel(const float* __restrict__ Z, const float* __restrict__ al,
                             const float* __restrict__ ar, float* __restrict__ el,
                             float* __restrict__ er) {
  int idx = blockIdx.x * blockDim.x + threadIdx.x;
  if (idx >= NT_ * HH) return;
  int n = idx / HH, h = idx - n * HH;
  const float* zr  = Z + (size_t)n * NF + h * D_;
  const float* alp = al + h * D_;
  const float* arp = ar + h * D_;
  float sl = 0.f, sr = 0.f;
  for (int d = 0; d < D_; ++d) {
    float zv = zr[d];
    sl += zv * alp[d];
    sr += zv * arp[d];
  }
  el[n * 8 + h] = sl;
  er[n * 8 + h] = sr;
}

__global__ void init_msd_kernel(float* __restrict__ m, float* __restrict__ den,
                                float* __restrict__ agg) {
  int idx = blockIdx.x * blockDim.x + threadIdx.x;
  if (idx < NT_ * 8) { m[idx] = -1e30f; den[idx] = 0.f; }
  if (idx < NT_ * 1500) agg[idx] = 0.f;
}

__global__ void edge_e_kernel(const int* __restrict__ src, const int* __restrict__ dst, int E,
                              const float* __restrict__ el, const float* __restrict__ er,
                              float* __restrict__ ebuf, float* __restrict__ m) {
  int idx = blockIdx.x * blockDim.x + threadIdx.x;
  if (idx >= E * HH) return;
  int e = idx / HH, h = idx - e * HH;
  float v = el[src[e] * 8 + h] + er[dst[e] * 8 + h];
  v = (v > 0.f) ? v : 0.2f * v;  // leaky_relu(0.2)
  ebuf[e * 8 + h] = v;
  atomic_fmax(&m[dst[e] * 8 + h], v);
}

__global__ void edge_ee_kernel(const int* __restrict__ dst, int E,
                               const float* __restrict__ m, float* __restrict__ ebuf,
                               float* __restrict__ den) {
  int idx = blockIdx.x * blockDim.x + threadIdx.x;
  if (idx >= E * HH) return;
  int e = idx / HH, h = idx - e * HH;
  float v = expf(ebuf[e * 8 + h] - m[dst[e] * 8 + h]);
  ebuf[e * 8 + h] = v;
  atomicAdd(&den[dst[e] * 8 + h], v);
}

__global__ void edge_agg_kernel(const int* __restrict__ src, const int* __restrict__ dst, int E,
                                const float* __restrict__ ebuf, const float* __restrict__ den,
                                const float* __restrict__ Z, float* __restrict__ agg) {
  int idx = blockIdx.x * blockDim.x + threadIdx.x;
  if (idx >= E * 1500) return;
  int e = idx / 1500;
  int c = idx - e * 1500;
  int h = c / D_;
  int de = dst[e];
  float alpha = ebuf[e * 8 + h] / den[de * 8 + h];
  atomicAdd(&agg[(size_t)de * 1500 + c], alpha * Z[(size_t)src[e] * NF + c]);
}

__global__ void combine_kernel(const float* __restrict__ agg, const float* __restrict__ R,
                               const float* __restrict__ gb, float* __restrict__ hf32,
                               _Float16* __restrict__ hf16) {
  int idx = blockIdx.x * blockDim.x + threadIdx.x;
  if (idx >= NT_ * D_) return;
  int n = idx / D_, d = idx - n * D_;
  float s = 0.f;
  for (int h = 0; h < HH; ++h)
    s += agg[(size_t)n * 1500 + h * D_ + d] + R[(size_t)n * NF + h * D_ + d] + gb[h * D_ + d];
  float hv = s * (1.f / HH);
  hf32[idx] = hv;
  hf16[(size_t)n * DP + d] = (_Float16)hv;
}

// ---------------- final pooling: context attention + output head ----------------
__global__ void __launch_bounds__(256)
final_kernel(const float* __restrict__ hf32, const float* __restrict__ py,
             const float* __restrict__ paW, const float* __restrict__ pab,
             const float* __restrict__ pac, const float* __restrict__ saW,
             const float* __restrict__ sab, const float* __restrict__ sac,
             const float* __restrict__ vW, const float* __restrict__ vb,
             const float* __restrict__ outW, const float* __restrict__ outb,
             float* __restrict__ out) {
  __shared__ float red[256];
  __shared__ float tk[64];
  __shared__ float pxs[304];
  __shared__ float sxs[304];
  int g = blockIdx.x, tid = threadIdx.x;

  // party attention (K = 8)
  for (int k = 0; k < PP; ++k) {
    const float* hn = hf32 + (size_t)(g * NODE_STRIDE + UU + k) * D_;
    float partial = 0.f;
    for (int d = tid; d < D_; d += 256) {
      float inner = 0.f;
      for (int dd = 0; dd < D_; ++dd) inner += hn[dd] * paW[dd * D_ + d];
      partial += tanhf(inner + pab[d]) * pac[d];
    }
    float tot = block_reduce_sum(partial, red);
    if (tid == 0) tk[k] = tot;
    __syncthreads();
  }
  if (tid == 0) {
    float mx = tk[0];
    for (int k = 1; k < PP; ++k) mx = fmaxf(mx, tk[k]);
    float s = 0.f;
    for (int k = 0; k < PP; ++k) { tk[k] = expf(tk[k] - mx); s += tk[k]; }
    for (int k = 0; k < PP; ++k) tk[k] /= s;
  }
  __syncthreads();
  for (int d = tid; d < D_; d += 256) {
    float s = 0.f;
    for (int k = 0; k < PP; ++k)
      s += tk[k] * hf32[(size_t)(g * NODE_STRIDE + UU + k) * D_ + d];
    pxs[d] = s;
  }
  __syncthreads();

  // sentence attention (K = 64)
  for (int k = 0; k < UU; ++k) {
    const float* hn = hf32 + (size_t)(g * NODE_STRIDE + k) * D_;
    float partial = 0.f;
    for (int d = tid; d < D_; d += 256) {
      float inner = 0.f;
      for (int dd = 0; dd < D_; ++dd) inner += hn[dd] * saW[dd * D_ + d];
      partial += tanhf(inner + sab[d]) * sac[d];
    }
    float tot = block_reduce_sum(partial, red);
    if (tid == 0) tk[k] = tot;
    __syncthreads();
  }
  if (tid == 0) {
    float mx = tk[0];
    for (int k = 1; k < UU; ++k) mx = fmaxf(mx, tk[k]);
    float s = 0.f;
    for (int k = 0; k < UU; ++k) { tk[k] = expf(tk[k] - mx); s += tk[k]; }
    for (int k = 0; k < UU; ++k) tk[k] /= s;
  }
  __syncthreads();
  for (int d = tid; d < D_; d += 256) {
    float s = 0.f;
    for (int k = 0; k < UU; ++k)
      s += tk[k] * hf32[(size_t)(g * NODE_STRIDE + k) * D_ + d];
    sxs[d] = s;
  }
  __syncthreads();

  float partial = 0.f;
  float pyg = py[g];
  for (int d = tid; d < D_; d += 256) {
    float vv = pyg * vW[d] + vb[d];
    partial += pxs[d] * outW[d] + sxs[d] * outW[D_ + d] + vv * outW[2 * D_ + d];
  }
  float tot = block_reduce_sum(partial, red);
  if (tid == 0) out[g] = tot + outb[0];
}

// ---------------- host orchestration ----------------
extern "C" void kernel_launch(void* const* d_in, const int* in_sizes, int n_in,
                              void* d_out, int out_size, void* d_ws, size_t ws_size,
                              hipStream_t stream) {
  const int*   x       = (const int*)d_in[0];
  const int*   src     = (const int*)d_in[1];
  const int*   dst     = (const int*)d_in[2];
  const int*   q_idx   = (const int*)d_in[4];
  const int*   pids    = (const int*)d_in[6];
  const float* py      = (const float*)d_in[7];
  const float* word_W  = (const float*)d_in[8];
  const float* w3      = (const float*)d_in[9];
  const float* b3      = (const float*)d_in[10];
  const float* w4      = (const float*)d_in[11];
  const float* b4      = (const float*)d_in[12];
  const float* w5      = (const float*)d_in[13];
  const float* b5      = (const float*)d_in[14];
  const float* party_W = (const float*)d_in[15];
  const float* pos_W   = (const float*)d_in[16];
  const float* gat_fc  = (const float*)d_in[17];
  const float* gat_al  = (const float*)d_in[18];
  const float* gat_ar  = (const float*)d_in[19];
  const float* gat_res = (const float*)d_in[20];
  const float* gat_b   = (const float*)d_in[21];
  const float* paW     = (const float*)d_in[22];
  const float* pab     = (const float*)d_in[23];
  const float* pac     = (const float*)d_in[24];
  const float* saW     = (const float*)d_in[25];
  const float* sab     = (const float*)d_in[26];
  const float* sac     = (const float*)d_in[27];
  const float* vW      = (const float*)d_in[28];
  const float* vb      = (const float*)d_in[29];
  const float* outW    = (const float*)d_in[30];
  const float* outb    = (const float*)d_in[31];
  float* out = (float*)d_out;
  const int E = in_sizes[1];

  char* ws = (char*)d_ws;
  size_t off = 0;
  auto alloc = [&](size_t bytes) -> void* {
    void* p = ws + off;
    off = (off + bytes + 255) & ~(size_t)255;
    return p;
  };
  _Float16* wfilt = (_Float16*)alloc((size_t)12 * CNT * KSTEPS * FRAG * 2);
  _Float16* gw    = (_Float16*)alloc((size_t)5 * 2 * GNT * KSTEPS * FRAG * 2);
  _Float16* hf16  = (_Float16*)alloc((size_t)NT_ * DP * 2);
  float*    hf32  = (float*)alloc((size_t)NT_ * D_ * 4);
  float*    Z     = (float*)alloc((size_t)NT_ * NF * 4);
  float*    Rb    = (float*)alloc((size_t)NT_ * NF * 4);
  float*    el    = (float*)alloc((size_t)NT_ * 8 * 4);
  float*    er    = (float*)alloc((size_t)NT_ * 8 * 4);
  float*    mb    = (float*)alloc((size_t)NT_ * 8 * 4);
  float*    den   = (float*)alloc((size_t)NT_ * 8 * 4);
  float*    ebuf  = (float*)alloc((size_t)E * 8 * 4);
  float*    agg   = (float*)alloc((size_t)NT_ * 1500 * 4);

  pack_filters_kernel<<<CDIV(12 * CNT * KSTEPS * FRAG, 256), 256, 0, stream>>>(w3, w4, w5, wfilt);
  pack_gatw_kernel<<<CDIV(5 * 2 * GNT * KSTEPS * FRAG, 256), 256, 0, stream>>>(gat_fc, gat_res, gw);
  zero_feat_kernel<<<CDIV(NT_ * DP, 256), 256, 0, stream>>>(hf32, hf16);
  party_fill_kernel<<<CDIV(BB * PP * D_, 256), 256, 0, stream>>>(q_idx, pids, party_W, hf32, hf16);

  size_t convsmem = (size_t)LR * DP * sizeof(_Float16) + 3 * 112 * sizeof(float);
  conv_embed_kernel<<<NU, 256, convsmem, stream>>>(x, word_W, wfilt, b3, b4, b5,
                                                   pos_W, hf32, hf16);

  const int mtiles = NT_ / 16;  // 144
  const int ntiles = GNT;       // 94
  const int gemm_blocks = CDIV(mtiles * ntiles, 4);
  for (int i = 0; i < 5; ++i) {
    wmma_gemm_kernel<<<gemm_blocks, 128, 0, stream>>>(
        hf16, gw + (size_t)(i * 2 + 0) * GNT * KSTEPS * FRAG, Z, mtiles, ntiles, KSTEPS, DP, NF);
    wmma_gemm_kernel<<<gemm_blocks, 128, 0, stream>>>(
        hf16, gw + (size_t)(i * 2 + 1) * GNT * KSTEPS * FRAG, Rb, mtiles, ntiles, KSTEPS, DP, NF);
    el_er_kernel<<<CDIV(NT_ * HH, 256), 256, 0, stream>>>(
        Z, gat_al + (size_t)i * 1500, gat_ar + (size_t)i * 1500, el, er);
    init_msd_kernel<<<CDIV(NT_ * 1500, 256), 256, 0, stream>>>(mb, den, agg);
    edge_e_kernel<<<CDIV(E * HH, 256), 256, 0, stream>>>(src, dst, E, el, er, ebuf, mb);
    edge_ee_kernel<<<CDIV(E * HH, 256), 256, 0, stream>>>(dst, E, mb, ebuf, den);
    edge_agg_kernel<<<CDIV(E * 1500, 256), 256, 0, stream>>>(src, dst, E, ebuf, den, Z, agg);
    combine_kernel<<<CDIV(NT_ * D_, 256), 256, 0, stream>>>(
        agg, Rb, gat_b + (size_t)i * 1500, hf32, hf16);
  }

  final_kernel<<<BB, 256, 0, stream>>>(hf32, py, paW, pab, pac, saW, sab, sac,
                                       vW, vb, outW, outb, out);
}
